// word_span_encoder_85710367359182
// MI455X (gfx1250) — matrix operand order repaired and checked by reading, not compiled, not run on hardware
//
#include <hip/hip_runtime.h>

// MI455X / gfx1250: wave32, WMMA 16x16x32 bf16 path.
// B=4096, L=64, D_IN=256, D_H=200 (padded to 208 for N-tiles, 256 for K).

typedef __attribute__((ext_vector_type(16))) __bf16       v16bf;
typedef __attribute__((ext_vector_type(8)))  float        v8f;
typedef __attribute__((ext_vector_type(4)))  unsigned int v4u;
typedef __attribute__((ext_vector_type(2)))  unsigned int v2u;
typedef __attribute__((ext_vector_type(4)))  float        v4f;

union Frag {
    v16bf        bf;
    v4u          q[2];
    unsigned int u[8];
};

__device__ __forceinline__ unsigned short f2bf(float f) {
    unsigned int u = __float_as_uint(f);
    u += 0x7FFFu + ((u >> 16) & 1u);        // round-to-nearest-even
    return (unsigned short)(u >> 16);
}

#define WMMA_BF16(A, Bm, C) \
    __builtin_amdgcn_wmma_f32_16x16x32_bf16(false, (A), false, (Bm), (short)0, (C), false, false)

// ---------------------------------------------------------------------------
// Prep: convert Wq/Wk/Wv ([200,256] -> bf16 [208,256], zero pad) and
// conv_w ([200,256,4] -> bf16 [208,1024], zero pad) into workspace.
// ---------------------------------------------------------------------------
#define QKV_ELEMS (3 * 208 * 256)          // 159744
#define CONV_ELEMS (208 * 1024)            // 212992
#define PREP_TOT (QKV_ELEMS + CONV_ELEMS)  // 372736

__global__ void wse_prep(const float* __restrict__ Wq, const float* __restrict__ Wk,
                         const float* __restrict__ Wv, const float* __restrict__ cw,
                         unsigned short* __restrict__ ws) {
    int idx = blockIdx.x * 256 + threadIdx.x;
    if (idx >= PREP_TOT) return;
    float val = 0.0f;
    if (idx < QKV_ELEMS) {
        int sel = idx / (208 * 256);
        int rem = idx - sel * (208 * 256);
        int row = rem >> 8, col = rem & 255;
        if (row < 200) {
            const float* W = (sel == 0) ? Wq : ((sel == 1) ? Wk : Wv);
            val = W[row * 256 + col];
        }
    } else {
        int rem = idx - QKV_ELEMS;
        int row = rem >> 10, col = rem & 1023;
        if (row < 200) val = cw[row * 1024 + col];
    }
    ws[idx] = f2bf(val);
}

// ---------------------------------------------------------------------------
// Main fused kernel: one workgroup (256 threads = 8 wave32) per batch item.
// ---------------------------------------------------------------------------
__global__ void __launch_bounds__(256)
wse_main(const float* __restrict__ x,
         const float* __restrict__ bq, const float* __restrict__ bk,
         const float* __restrict__ bv, const float* __restrict__ cb,
         const unsigned short* __restrict__ wsA,   // 3 x [208][256] bf16
         const unsigned short* __restrict__ wsC,   // [208][1024] bf16
         float* __restrict__ out) {
    __shared__ __align__(16) unsigned short xb [64 * 256];  // x (bf16), also conv input (flat)
    __shared__ __align__(16) unsigned short qs [64 * 256];  // q, K-padded with zeros
    __shared__ __align__(16) unsigned short ksm[64 * 256];  // k, K-padded with zeros
    __shared__ __align__(16) unsigned short vT [256 * 64];  // v transposed [h][m]
    __shared__ __align__(16) unsigned short sm [64 * 64];   // softmax(scores), bf16
    __shared__ __align__(16) float          scpool[64 * 64];// scores f32; later out pool [16*200]

    const int b     = blockIdx.x;
    const int tid   = threadIdx.x;
    const int wave  = tid >> 5;
    const int lane  = tid & 31;
    const int ln16  = lane & 15;
    const int khalf = lane >> 4;

    // ---- Phase 0: x -> LDS bf16; zero q/k/vT pads ----
    const float* xg = x + (size_t)b * 16384;
    for (int i = tid * 4; i < 16384; i += 1024) {
        v4f v = ((const v4f*)xg)[i >> 2];
        v2u pk;
        pk.x = (unsigned)f2bf(v.x) | ((unsigned)f2bf(v.y) << 16);
        pk.y = (unsigned)f2bf(v.z) | ((unsigned)f2bf(v.w) << 16);
        *(v2u*)(xb + i) = pk;
    }
    {
        unsigned int* z0 = (unsigned int*)qs;
        unsigned int* z1 = (unsigned int*)ksm;
        unsigned int* z2 = (unsigned int*)vT;
        for (int i = tid; i < 8192; i += 256) { z0[i] = 0u; z1[i] = 0u; z2[i] = 0u; }
    }
    __syncthreads();

    // ---- Phase 1: QKV projections (156 tiles: 3 mats x 4 Mtiles x 13 Ntiles) ----
    for (int t = wave; t < 156; t += 8) {
        int mat = t / 52;
        int rem = t - mat * 52;
        int mt  = rem / 13;
        int nt  = rem - mt * 13;
        int m0 = mt << 4, n0 = nt << 4;
        const unsigned short* Arow = xb  + (m0 + ln16) * 256;
        const unsigned short* Brow = wsA + mat * 53248 + (n0 + ln16) * 256;
        v8f acc = {};
#pragma unroll
        for (int k0 = 0; k0 < 256; k0 += 32) {
            Frag a, bm;
            a.q[0]  = *(const v4u*)(Arow + k0 + 8 * khalf);
            a.q[1]  = *(const v4u*)(Arow + k0 + 16 + 8 * khalf);
            bm.q[0] = *(const v4u*)(Brow + k0 + 8 * khalf);
            bm.q[1] = *(const v4u*)(Brow + k0 + 16 + 8 * khalf);
            acc = WMMA_BF16(a.bf, bm.bf, acc);
        }
        int n = n0 + ln16;
        const float* bp = (mat == 0) ? bq : ((mat == 1) ? bk : bv);
        float bias = (n < 200) ? bp[n] : 0.0f;
        if (mat == 2) {   // v stored transposed: vT[h][m], 8 contiguous bf16 per lane
            v4u pk;
            unsigned short s0, s1;
            float e;
            e = acc[0] + bias; e = e > 0.f ? e : 0.f; s0 = f2bf(e);
            e = acc[1] + bias; e = e > 0.f ? e : 0.f; s1 = f2bf(e);
            pk.x = (unsigned)s0 | ((unsigned)s1 << 16);
            e = acc[2] + bias; e = e > 0.f ? e : 0.f; s0 = f2bf(e);
            e = acc[3] + bias; e = e > 0.f ? e : 0.f; s1 = f2bf(e);
            pk.y = (unsigned)s0 | ((unsigned)s1 << 16);
            e = acc[4] + bias; e = e > 0.f ? e : 0.f; s0 = f2bf(e);
            e = acc[5] + bias; e = e > 0.f ? e : 0.f; s1 = f2bf(e);
            pk.z = (unsigned)s0 | ((unsigned)s1 << 16);
            e = acc[6] + bias; e = e > 0.f ? e : 0.f; s0 = f2bf(e);
            e = acc[7] + bias; e = e > 0.f ? e : 0.f; s1 = f2bf(e);
            pk.w = (unsigned)s0 | ((unsigned)s1 << 16);
            *(v4u*)(vT + n * 64 + m0 + 8 * khalf) = pk;
        } else {
            unsigned short* dst = (mat == 0) ? qs : ksm;
#pragma unroll
            for (int r = 0; r < 8; r++) {
                float e = acc[r] + bias; e = e > 0.f ? e : 0.f;
                dst[(m0 + 8 * khalf + r) * 256 + n] = f2bf(e);
            }
        }
    }
    __syncthreads();

    // ---- Phase 2: scores[l][m] = q[l,:] . k[m,:]  (16 tiles, K=256) ----
    for (int t = wave; t < 16; t += 8) {
        int m0 = (t >> 2) << 4, n0 = (t & 3) << 4;
        const unsigned short* Arow = qs  + (m0 + ln16) * 256;
        const unsigned short* Brow = ksm + (n0 + ln16) * 256;
        v8f acc = {};
#pragma unroll
        for (int k0 = 0; k0 < 256; k0 += 32) {
            Frag a, bm;
            a.q[0]  = *(const v4u*)(Arow + k0 + 8 * khalf);
            a.q[1]  = *(const v4u*)(Arow + k0 + 16 + 8 * khalf);
            bm.q[0] = *(const v4u*)(Brow + k0 + 8 * khalf);
            bm.q[1] = *(const v4u*)(Brow + k0 + 16 + 8 * khalf);
            acc = WMMA_BF16(a.bf, bm.bf, acc);
        }
        int n = n0 + ln16;
#pragma unroll
        for (int r = 0; r < 8; r++)
            scpool[(m0 + 8 * khalf + r) * 64 + n] = acc[r];
    }
    __syncthreads();

    // ---- Phase 3: softmax over l (axis=1) per column m ----
    if (tid < 64) {
        int m = tid;
        float mx = -3.4e38f;
        for (int l = 0; l < 64; l++) { float v = scpool[l * 64 + m]; mx = v > mx ? v : mx; }
        float s = 0.f;
        for (int l = 0; l < 64; l++) s += __expf(scpool[l * 64 + m] - mx);
        float inv = 1.0f / s;
        for (int l = 0; l < 64; l++)
            sm[l * 64 + m] = f2bf(__expf(scpool[l * 64 + m] - mx) * inv);
    }
    __syncthreads();

    // reuse scpool as the fused [16,200] output accumulator
    float* outp = scpool;
    for (int i = tid; i < 3200; i += 256) outp[i] = 0.0f;
    __syncthreads();

    // ---- Phase 4: self_att = soft @ v, fused 4-way mean-pool into outp ----
    for (int t = wave; t < 52; t += 8) {
        int m0 = (t / 13) << 4, n0 = (t % 13) << 4;
        const unsigned short* Arow = sm + (m0 + ln16) * 64;
        const unsigned short* Brow = vT + (n0 + ln16) * 64;
        v8f acc = {};
#pragma unroll
        for (int k0 = 0; k0 < 64; k0 += 32) {
            Frag a, bm;
            a.q[0]  = *(const v4u*)(Arow + k0 + 8 * khalf);
            a.q[1]  = *(const v4u*)(Arow + k0 + 16 + 8 * khalf);
            bm.q[0] = *(const v4u*)(Brow + k0 + 8 * khalf);
            bm.q[1] = *(const v4u*)(Brow + k0 + 16 + 8 * khalf);
            acc = WMMA_BF16(a.bf, bm.bf, acc);
        }
        int h = n0 + ln16;
        if (h < 200) {
#pragma unroll
            for (int r = 0; r < 8; r++) {
                int l = m0 + 8 * khalf + r;
                int f = l * 200 + h;                       // flat [64,200]
                int idx = ((f >> 6) << 4) | (f & 15);      // mean-pool reshape target
                atomicAdd(&outp[idx], acc[r] * 0.25f);
            }
        }
    }

    // ---- Phase 5: conv as GEMM  A[16,1024] x conv_w^T[1024,208] ----
    for (int t = wave; t < 13; t += 8) {
        int n0 = t << 4;
        int o  = n0 + ln16;                 // output channel (padded rows are zero)
        const unsigned short* Brow = wsC + o * 1024;
        int j = ln16;                       // A-matrix row = output position
        v8f acc = {};
        for (int k0 = 0; k0 < 1024; k0 += 32) {
            Frag a, bm;
#pragma unroll
            for (int r = 0; r < 8; r++) {
                int kb   = k0 + ((r >= 4) ? 16 : 0) + 8 * khalf + 2 * (r & 3);
                int flat = ((kb >> 2) << 6) + 4 * j + (kb & 3);  // x_flat[i*64+4j+kk]
                a.u[r]   = *(const unsigned int*)(xb + flat);    // pair (kb, kb+1)
            }
            bm.q[0] = *(const v4u*)(Brow + k0 + 8 * khalf);
            bm.q[1] = *(const v4u*)(Brow + k0 + 16 + 8 * khalf);
            acc = WMMA_BF16(a.bf, bm.bf, acc);
        }
        if (o < 200) {
            float cbias = cb[o];
#pragma unroll
            for (int r = 0; r < 8; r++) {
                int jj = 8 * khalf + r;
                atomicAdd(&outp[o * 16 + jj], acc[r] + cbias);   // conv reshape is flat-identity
            }
        }
    }
    __syncthreads();

    // ---- Phase 6: writeout [16,200] ----
    float* og = out + (size_t)b * 3200;
    for (int i = tid; i < 3200; i += 256) og[i] = outp[i];
}

// ---------------------------------------------------------------------------
extern "C" void kernel_launch(void* const* d_in, const int* in_sizes, int n_in,
                              void* d_out, int out_size, void* d_ws, size_t ws_size,
                              hipStream_t stream) {
    (void)in_sizes; (void)n_in; (void)out_size; (void)ws_size;
    const float* x  = (const float*)d_in[0];
    const float* Wq = (const float*)d_in[1];
    const float* bq = (const float*)d_in[2];
    const float* Wk = (const float*)d_in[3];
    const float* bk = (const float*)d_in[4];
    const float* Wv = (const float*)d_in[5];
    const float* bv = (const float*)d_in[6];
    const float* cw = (const float*)d_in[7];
    const float* cb = (const float*)d_in[8];
    unsigned short* wsA = (unsigned short*)d_ws;
    unsigned short* wsC = wsA + QKV_ELEMS;
    float* out = (float*)d_out;

    wse_prep<<<PREP_TOT / 256, 256, 0, stream>>>(Wq, Wk, Wv, cw, wsA);
    wse_main<<<4096, 256, 0, stream>>>(x, bq, bk, bv, cb, wsA, wsC, out);
}
